// SelfAttention_6579889897578
// MI455X (gfx1250) — compile-verified, matrix operand order checked by reading, not run on hardware
//
#include <hip/hip_runtime.h>
#include <hip/hip_bf16.h>

typedef __bf16 bf16;
typedef __attribute__((ext_vector_type(16))) __bf16 v16bf;
typedef __attribute__((ext_vector_type(8)))  float  v8f;

enum : int { B_ = 4, S_ = 4096, D_ = 512, FF_ = 2048, BS_ = 16384, BAND_ = 160 };

#define WMMA_BF16(a, b, c) \
  __builtin_amdgcn_wmma_f32_16x16x32_bf16(false, (a), false, (b), (short)0, (c), false, false)

// Load one 16x32 bf16 fragment (A from row-major MxK, or B from "N-major" NxK,
// i.e. operand stored as [row/col][k]). Caller passes pointer to this lane's
// row at k-offset 0 of the fragment. K layout per CDNA5 ISA 7.12.2:
// lanes 0-15: K = {0..7} u {16..23}; lanes 16-31: K = {8..15} u {24..31}.
__device__ __forceinline__ v16bf ldfrag(const bf16* rowptr, int hi) {
  union { v16bf v; unsigned u[8]; } f;
  const unsigned* pr = (const unsigned*)rowptr;
#pragma unroll
  for (int q = 0; q < 8; ++q) {
    int kb = ((q & 4) ? 16 : 0) + (hi ? 8 : 0) + ((q & 3) << 1);
    f.u[q] = pr[kb >> 1];
  }
  return f.v;
}

// B fragment for attn@V: operand stored as srcT[d][s] (contiguous in k = seq),
// with per-dword clamp of the sequence index into [0, Slim-2] (clamped lanes
// multiply zero probabilities, so values are don't-care; clamp is for safety).
__device__ __forceinline__ v16bf ldfrag_vk(const bf16* colptr, int hi, int k0, int Slim) {
  union { v16bf v; unsigned u[8]; } f;
#pragma unroll
  for (int q = 0; q < 8; ++q) {
    int kb = ((q & 4) ? 16 : 0) + (hi ? 8 : 0) + ((q & 3) << 1);
    int s = k0 + kb;
    s = s < 0 ? 0 : (s > Slim - 2 ? Slim - 2 : s);
    f.u[q] = *(const unsigned*)(colptr + s);
  }
  return f.v;
}

// ---------------- casts ----------------
__global__ void k_cast(const float* __restrict__ in, bf16* __restrict__ out, int n) {
  int i = blockIdx.x * 256 + threadIdx.x;
  if (i < n) out[i] = (bf16)in[i];
}

__global__ void k_cast_src(const float* __restrict__ src, bf16* __restrict__ srch,
                           bf16* __restrict__ srcT) {
  int i = blockIdx.x * 256 + threadIdx.x;
  if (i < BS_ * D_) {
    int s = i >> 9, d = i & 511;
    float v = src[i];
    srch[i] = (bf16)v;
    srcT[(size_t)d * BS_ + s] = (bf16)v;
  }
}

// ---------------- K1: banded scores + softmax -> probs ----------------
// grid = 1024 tiles (16 rows each), block = 64 threads (2 waves, 5 col-tiles each)
__global__ void k_scores_softmax(const bf16* __restrict__ srch, bf16* __restrict__ probs) {
  const int t = blockIdx.x;
  const int b = t >> 8;                 // 256 tiles per batch
  const int jl0 = (t & 255) << 4;       // batch-local row base
  const int colbase = jl0 - 64;         // batch-local col base (may be <0)
  const int wav = threadIdx.x >> 5;
  const int lane = threadIdx.x & 31;
  const int hi = lane >> 4, ln = lane & 15;
  __shared__ float sc[16][BAND_];
  __shared__ bf16 qs[16][D_];           // staged Q tile (16 KB)

  // stage Q tile once (avoids 5x refetch across col-tiles)
  {
    const uint4* sp = (const uint4*)(srch + (size_t)(b * S_ + jl0) * D_);
    uint4* dp = (uint4*)&qs[0][0];
    for (int e = threadIdx.x; e < 16 * D_ * 2 / 16; e += 64) dp[e] = sp[e];
  }
  __syncthreads();

  const float scale = 0.044194173824159216f;  // 1/sqrt(512)

  for (int ct = wav * 5; ct < wav * 5 + 5; ++ct) {
    int cl = colbase + ct * 16 + ln;
    int clc = cl < 0 ? 0 : (cl >= S_ ? S_ - 1 : cl);   // clamp; masked in softmax
    const bf16* krow = srch + (size_t)(b * S_ + clc) * D_;
    v8f acc = {};
    for (int kc = 0; kc < D_; kc += 32) {
      v16bf a = ldfrag(&qs[ln][kc], hi);
      v16bf bm = ldfrag(krow + kc, hi);
      acc = WMMA_BF16(a, bm, acc);
    }
#pragma unroll
    for (int r = 0; r < 8; ++r) sc[r + hi * 8][ct * 16 + ln] = acc[r] * scale;
  }
  __syncthreads();

  if (threadIdx.x < 16) {
    int r = threadIdx.x;
    float mx = -1e30f;
    for (int c = 0; c < BAND_; ++c) {
      int cl = colbase + c;
      bool valid = (c >= r) && (c <= r + 128) && (cl >= 0) && (cl < S_);
      if (valid) { float v = sc[r][c]; mx = v > mx ? v : mx; }
    }
    float sum = 0.f;
    for (int c = 0; c < BAND_; ++c) {
      int cl = colbase + c;
      bool valid = (c >= r) && (c <= r + 128) && (cl >= 0) && (cl < S_);
      float e = valid ? __expf(sc[r][c] - mx) : 0.f;
      sc[r][c] = e;
      sum += e;
    }
    float rs = 1.f / sum;
    bf16* prow = probs + (size_t)t * 16 * BAND_ + r * BAND_;
    for (int c = 0; c < BAND_; ++c) prow[c] = (bf16)(sc[r][c] * rs);
  }
}

// ---------------- K2: attn @ V, fused ReLU + bf16 cast ----------------
// grid = 1024 tiles, block = 128 (4 waves x 8 d-tiles)
__global__ void k_attn_v(const bf16* __restrict__ probs, const bf16* __restrict__ srcT,
                         bf16* __restrict__ ar) {
  const int t = blockIdx.x;
  const int b = t >> 8;
  const int jl0 = (t & 255) << 4;
  const int colbase = jl0 - 64;
  const int wav = threadIdx.x >> 5;
  const int lane = threadIdx.x & 31;
  const int hi = lane >> 4, ln = lane & 15;
  __shared__ bf16 ps[16][BAND_];        // staged prob tile (5 KB)

  {
    const uint4* sp = (const uint4*)(probs + (size_t)t * 16 * BAND_);
    uint4* dp = (uint4*)&ps[0][0];
    for (int e = threadIdx.x; e < 16 * BAND_ * 2 / 16; e += 128) dp[e] = sp[e];
  }
  __syncthreads();

  v8f acc[8];
#pragma unroll
  for (int i = 0; i < 8; ++i) acc[i] = (v8f){};

  for (int kc = 0; kc < BAND_; kc += 32) {
    v16bf a = ldfrag(&ps[ln][kc], hi);
#pragma unroll
    for (int i = 0; i < 8; ++i) {
      int d0 = (wav * 8 + i) * 16;
      const bf16* col = srcT + (size_t)(d0 + ln) * BS_ + b * S_;
      v16bf bm = ldfrag_vk(col, hi, colbase + kc, S_);
      acc[i] = WMMA_BF16(a, bm, acc[i]);
    }
  }
  const int g0 = t * 16;
#pragma unroll
  for (int i = 0; i < 8; ++i) {
    int d0 = (wav * 8 + i) * 16;
#pragma unroll
    for (int r = 0; r < 8; ++r) {
      int m = r + hi * 8;
      float v = acc[i][r];
      v = v > 0.f ? v : 0.f;
      ar[(size_t)(g0 + m) * D_ + d0 + ln] = (bf16)v;
    }
  }
}

// ---------------- K3: conv(1x1) GEMM + bias + residual + LayerNorm1 ----------------
__global__ void k_conv_ln1(const bf16* __restrict__ ar, const bf16* __restrict__ cwh,
                           const float* __restrict__ conv_b, const float* __restrict__ src,
                           const float* __restrict__ g1, const float* __restrict__ be1,
                           float* __restrict__ x, bf16* __restrict__ xh) {
  const int t = blockIdx.x;
  const int g0 = t * 16;
  const int wav = threadIdx.x >> 5;
  const int lane = threadIdx.x & 31;
  const int hi = lane >> 4, ln = lane & 15;
  __shared__ float xs[16][D_];
  __shared__ bf16 as[16][D_];           // staged activation tile (16 KB)
  __shared__ float red[16][8][2];
  __shared__ float stats[16][2];

  if (threadIdx.x < 32) __builtin_prefetch(cwh + threadIdx.x * 64, 0, 1);
  {
    const uint4* sp = (const uint4*)(ar + (size_t)g0 * D_);
    uint4* dp = (uint4*)&as[0][0];
    for (int e = threadIdx.x; e < 16 * D_ * 2 / 16; e += 128) dp[e] = sp[e];
  }
  __syncthreads();

  v8f acc[8];
#pragma unroll
  for (int i = 0; i < 8; ++i) acc[i] = (v8f){};

  for (int kc = 0; kc < D_; kc += 32) {
    v16bf a = ldfrag(&as[ln][kc], hi);
#pragma unroll
    for (int i = 0; i < 8; ++i) {
      int d0 = (wav * 8 + i) * 16;
      v16bf bm = ldfrag(cwh + (size_t)(d0 + ln) * D_ + kc, hi);
      acc[i] = WMMA_BF16(a, bm, acc[i]);
    }
  }
#pragma unroll
  for (int i = 0; i < 8; ++i) {
    int n = (wav * 8 + i) * 16 + ln;
#pragma unroll
    for (int r = 0; r < 8; ++r) {
      int m = r + hi * 8;
      xs[m][n] = acc[i][r] + conv_b[n] + src[(size_t)(g0 + m) * D_ + n];
    }
  }
  __syncthreads();
  {
    int r = threadIdx.x & 15, part = threadIdx.x >> 4;
    float s = 0.f, s2 = 0.f;
    for (int c = part * 64; c < part * 64 + 64; ++c) { float v = xs[r][c]; s += v; s2 += v * v; }
    red[r][part][0] = s; red[r][part][1] = s2;
  }
  __syncthreads();
  if (threadIdx.x < 16) {
    int r = threadIdx.x;
    float s = 0.f, s2 = 0.f;
    for (int p = 0; p < 8; ++p) { s += red[r][p][0]; s2 += red[r][p][1]; }
    float mu = s * (1.f / D_);
    float var = s2 * (1.f / D_) - mu * mu;
    stats[r][0] = mu;
    stats[r][1] = rsqrtf(var + 1e-5f);
  }
  __syncthreads();
  for (int e = threadIdx.x; e < 16 * D_; e += 128) {
    int m = e >> 9, c = e & 511;
    float v = (xs[m][c] - stats[m][0]) * stats[m][1] * g1[c] + be1[c];
    x[(size_t)(g0 + m) * D_ + c] = v;
    xh[(size_t)(g0 + m) * D_ + c] = (bf16)v;
  }
}

// ---------------- K4: fused FFN (h kept in LDS) + residual + LayerNorm2 ----------------
__global__ void k_ffn_ln2(const bf16* __restrict__ xh, const float* __restrict__ x,
                          const bf16* __restrict__ w1h, const float* __restrict__ b1,
                          const bf16* __restrict__ w2h, const float* __restrict__ b2,
                          const float* __restrict__ g2, const float* __restrict__ be2,
                          float* __restrict__ out) {
  const int t = blockIdx.x;
  const int g0 = t * 16;
  const int wav = threadIdx.x >> 5;
  const int lane = threadIdx.x & 31;
  const int hi = lane >> 4, ln = lane & 15;

  union ShBuf { bf16 h[16][1024]; float o[16][D_]; };
  __shared__ ShBuf sh;
  __shared__ bf16 xsh[16][D_];          // staged x tile (16 KB), reused by 2x64 n-tiles
  __shared__ float red[16][8][2];
  __shared__ float stats[16][2];

  if (threadIdx.x < 32) {
    __builtin_prefetch(w1h + threadIdx.x * 64, 0, 1);
    __builtin_prefetch(w2h + threadIdx.x * 64, 0, 1);
  }
  {
    const uint4* sp = (const uint4*)(xh + (size_t)g0 * D_);
    uint4* dp = (uint4*)&xsh[0][0];
    for (int e = threadIdx.x; e < 16 * D_ * 2 / 16; e += 128) dp[e] = sp[e];
  }
  __syncthreads();

  v8f accum[8];
#pragma unroll
  for (int i = 0; i < 8; ++i) accum[i] = (v8f){};

  for (int half = 0; half < 2; ++half) {
    const int fbase = half * 1024;
    // GEMM1: h[:, fbase:fbase+1024] = relu(x @ w1^T + b1), bf16 into LDS
    for (int i = 0; i < 16; ++i) {
      int fl = (wav * 16 + i) * 16;  // local column base within half
      v8f a1 = {};
      for (int kc = 0; kc < D_; kc += 32) {
        v16bf a = ldfrag(&xsh[ln][kc], hi);
        v16bf bm = ldfrag(w1h + (size_t)(fbase + fl + ln) * D_ + kc, hi);
        a1 = WMMA_BF16(a, bm, a1);
      }
#pragma unroll
      for (int r = 0; r < 8; ++r) {
        int m = r + hi * 8;
        float v = a1[r] + b1[fbase + fl + ln];
        v = v > 0.f ? v : 0.f;
        sh.h[m][fl + ln] = (bf16)v;
      }
    }
    __syncthreads();
    // GEMM2 partial accumulation over this half's K = 1024
    for (int i = 0; i < 8; ++i) {
      int d0 = (wav * 8 + i) * 16;
      for (int kc = 0; kc < 1024; kc += 32) {
        v16bf a = ldfrag(&sh.h[ln][kc], hi);
        v16bf bm = ldfrag(w2h + (size_t)(d0 + ln) * FF_ + fbase + kc, hi);
        accum[i] = WMMA_BF16(a, bm, accum[i]);
      }
    }
    __syncthreads();  // h reused (half 2) or overlaid by output buffer
  }

#pragma unroll
  for (int i = 0; i < 8; ++i) {
    int n = (wav * 8 + i) * 16 + ln;
#pragma unroll
    for (int r = 0; r < 8; ++r) {
      int m = r + hi * 8;
      sh.o[m][n] = accum[i][r] + b2[n] + x[(size_t)(g0 + m) * D_ + n];
    }
  }
  __syncthreads();
  {
    int r = threadIdx.x & 15, part = threadIdx.x >> 4;
    float s = 0.f, s2 = 0.f;
    for (int c = part * 64; c < part * 64 + 64; ++c) { float v = sh.o[r][c]; s += v; s2 += v * v; }
    red[r][part][0] = s; red[r][part][1] = s2;
  }
  __syncthreads();
  if (threadIdx.x < 16) {
    int r = threadIdx.x;
    float s = 0.f, s2 = 0.f;
    for (int p = 0; p < 8; ++p) { s += red[r][p][0]; s2 += red[r][p][1]; }
    float mu = s * (1.f / D_);
    float var = s2 * (1.f / D_) - mu * mu;
    stats[r][0] = mu;
    stats[r][1] = rsqrtf(var + 1e-5f);
  }
  __syncthreads();
  for (int e = threadIdx.x; e < 16 * D_; e += 128) {
    int m = e >> 9, c = e & 511;
    float v = (sh.o[m][c] - stats[m][0]) * stats[m][1] * g2[c] + be2[c];
    out[(size_t)(g0 + m) * D_ + c] = v;
  }
}

extern "C" void kernel_launch(void* const* d_in, const int* in_sizes, int n_in,
                              void* d_out, int out_size, void* d_ws, size_t ws_size,
                              hipStream_t stream) {
  (void)in_sizes; (void)n_in; (void)out_size; (void)ws_size;
  const float* src    = (const float*)d_in[0];
  const float* conv_w = (const float*)d_in[1];
  const float* conv_b = (const float*)d_in[2];
  const float* w1     = (const float*)d_in[3];
  const float* b1     = (const float*)d_in[4];
  const float* w2     = (const float*)d_in[5];
  const float* b2     = (const float*)d_in[6];
  const float* g1     = (const float*)d_in[7];
  const float* be1    = (const float*)d_in[8];
  const float* g2     = (const float*)d_in[9];
  const float* be2    = (const float*)d_in[10];
  float* out = (float*)d_out;

  char* p = (char*)d_ws;
  auto alloc = [&](size_t bytes) {
    char* r = p;
    p += (bytes + 255) & ~(size_t)255;
    return r;
  };
  bf16* srch  = (bf16*)alloc((size_t)BS_ * D_ * 2);
  bf16* srcT  = (bf16*)alloc((size_t)BS_ * D_ * 2);
  bf16* cwh   = (bf16*)alloc((size_t)D_ * D_ * 2);
  bf16* w1h   = (bf16*)alloc((size_t)FF_ * D_ * 2);
  bf16* w2h   = (bf16*)alloc((size_t)D_ * FF_ * 2);
  bf16* probs = (bf16*)alloc((size_t)(BS_ / 16) * 16 * BAND_ * 2);
  bf16* ar    = (bf16*)alloc((size_t)BS_ * D_ * 2);
  float* x    = (float*)alloc((size_t)BS_ * D_ * 4);
  bf16* xh    = (bf16*)alloc((size_t)BS_ * D_ * 2);

  k_cast_src<<<(BS_ * D_ + 255) / 256, 256, 0, stream>>>(src, srch, srcT);
  k_cast<<<(D_ * D_ + 255) / 256, 256, 0, stream>>>(conv_w, cwh, D_ * D_);
  k_cast<<<(FF_ * D_ + 255) / 256, 256, 0, stream>>>(w1, w1h, FF_ * D_);
  k_cast<<<(D_ * FF_ + 255) / 256, 256, 0, stream>>>(w2, w2h, D_ * FF_);

  const int NT = BS_ / 16;  // 1024 row tiles
  k_scores_softmax<<<NT, 64, 0, stream>>>(srch, probs);
  k_attn_v<<<NT, 128, 0, stream>>>(probs, srcT, ar);
  k_conv_ln1<<<NT, 128, 0, stream>>>(ar, cwh, conv_b, src, g1, be1, x, xh);
  k_ffn_ln2<<<NT, 128, 0, stream>>>(xh, x, w1h, b1, w2h, b2, g2, be2, out);
}
